// Decoder_74509092651036
// MI455X (gfx1250) — 8-bit matrix operand order repaired and checked by reading, compiled, not run on hardware
//
#include <hip/hip_runtime.h>
#include <math.h>


#define T_LEN    200
#define V_LEN    29
#define TPAD     256
#define TW       (TPAD / 4)
#define BMAX     10
#define CMAX     (2 * BMAX * BMAX)
#define RMAXP    112
#define EQW      ((RMAXP + 31) / 32)
#define NTHREADS 256
#define NEGF     (-1e30f)
#define DUPF     (-2e30f)
#define MINF     (-3.0e38f)

typedef int v8i __attribute__((ext_vector_type(8)));

__device__ __forceinline__ float laddexp(float a, float b) {
  float m = fmaxf(a, b);
  float n = fminf(a, b);
  return m + log1pf(__expf(n - m));
}

__global__ __launch_bounds__(NTHREADS)
void ctc_prefix_beam_kernel(const float* __restrict__ logits,
                            const int* __restrict__ p_beam,
                            const int* __restrict__ p_blank,
                            float* __restrict__ out)
{
  __shared__ signed char  cur[BMAX][TPAD];
  __shared__ signed char  rows[RMAXP][TPAD];
  __shared__ unsigned int eqm[RMAXP][EQW];
  __shared__ float logp[V_LEN];
  __shared__ float ps[BMAX];
  __shared__ int   ssym[BMAX];
  __shared__ float beam_pb[BMAX], beam_pnb[BMAX];
  __shared__ int   beam_len[BMAX], lastv[BMAX];
  __shared__ float cpb[CMAX], cpnb[CMAX];
  __shared__ int   clen[CMAX];
  __shared__ int   norms[RMAXP];
  __shared__ float mpb[CMAX], mpnb[CMAX], keyv[CMAX];
  __shared__ int   sel[BMAX];

  const int tid = threadIdx.x;
  int B = p_beam[0]; if (B < 1) B = 1; if (B > BMAX) B = BMAX;
  const int blank = p_blank[0];
  const int Kc = B;
  const int C  = 2 * B * Kc;
  const int R  = B + B * Kc;
  const int RP = (R + 15) & ~15;
  const int TILES = RP >> 4;

  for (int i = tid; i < BMAX * TPAD; i += NTHREADS) {
    int b = i / TPAD, t = i % TPAD;
    signed char v = -1;
    if (t == 0 && b > 0) v = (signed char)(-(b + 2));
    cur[b][t] = v;
  }
  for (int i = tid; i < RMAXP * TPAD; i += NTHREADS)
    rows[i / TPAD][i % TPAD] = 0;
  if (tid < BMAX) {
    beam_pb[tid]  = (tid == 0) ? 0.0f : NEGF;
    beam_pnb[tid] = NEGF;
    beam_len[tid] = 0;
  }
  __syncthreads();

  for (int t = 0; t < T_LEN; ++t) {
    if (tid == 0) {
      const float* x = logits + t * V_LEN;
      float mx = x[0];
      for (int v = 1; v < V_LEN; ++v) mx = fmaxf(mx, x[v]);
      float sum = 0.0f;
      for (int v = 0; v < V_LEN; ++v) sum += __expf(x[v] - mx);
      float lse = mx + __logf(sum);
      for (int v = 0; v < V_LEN; ++v) logp[v] = x[v] - lse;
      unsigned int used = 0u;
      for (int k = 0; k < Kc; ++k) {
        float best = MINF; int bi = 0;
        for (int v = 0; v < V_LEN; ++v)
          if (!((used >> v) & 1u) && logp[v] > best) { best = logp[v]; bi = v; }
        used |= 1u << bi;
        ps[k] = best; ssym[k] = bi;
      }
    }
    if (tid < B) {
      int len = beam_len[tid];
      lastv[tid] = (len > 0) ? (int)cur[tid][len - 1] : -1;
    }
    for (int i = tid; i < RMAXP * EQW; i += NTHREADS)
      ((unsigned int*)eqm)[i] = 0u;
    __syncthreads();

    for (int r = tid; r < R; r += NTHREADS) {
      int b = (r < B) ? r : (r - B) / Kc;
      int* dst = (int*)rows[r];
      const int* src = (const int*)cur[b];
      for (int w = 0; w < TW; ++w) dst[w] = src[w];
      if (r >= B) {
        int k = (r - B) % Kc;
        rows[r][beam_len[b]] = (signed char)ssym[k];
      }
    }
    for (int c = tid; c < C; c += NTHREADS) {
      int g = c / (B * Kc);
      int rem = c % (B * Kc);
      int b = rem / Kc, k = rem % Kc;
      float psk = ps[k]; int sk = ssym[k];
      bool isb  = (sk == blank);
      bool same = (lastv[b] == sk);
      float pb_ps  = beam_pb[b] + psk;
      float pnb_ps = beam_pnb[b] + psk;
      float both   = laddexp(pb_ps, pnb_ps);
      if (g == 0) {
        cpb[c]  = isb ? both : NEGF;
        cpnb[c] = (same && !isb) ? pnb_ps : NEGF;
        clen[c] = beam_len[b];
      } else {
        cpb[c]  = NEGF;
        cpnb[c] = isb ? NEGF : (same ? pb_ps : both);
        clen[c] = beam_len[b] + 1;
      }
    }
    __syncthreads();

    for (int r = tid; r < RP; r += NTHREADS) {
      int acc = 0;
      for (int w = 0; w < TW; ++w) {
        int word = ((const int*)rows[r])[w];
        #pragma unroll
        for (int by = 0; by < 4; ++by) {
          int e = (int)(signed char)(word >> (8 * by));
          acc += e * e;
        }
      }
      norms[r] = acc;
    }
    __syncthreads();

    {
      const int wave = tid >> 5;
      const int lane = tid & 31;
      const int m15  = lane & 15;
      const int hf   = lane >> 4;
      const int npairs = TILES * TILES;
      for (int p = wave; p < npairs; p += NTHREADS / 32) {
        const int ti = p / TILES, tj = p % TILES;
        const int ra = ti * 16 + m15;
        const int rb = tj * 16 + m15;
        v8i acc = {0, 0, 0, 0, 0, 0, 0, 0};
        for (int ch = 0; ch < TPAD / 64; ++ch) {
          v8i av, bv;
          const int kb = ch * 64;
          #pragma unroll
          for (int v = 0; v < 8; ++v) {
            int ka = kb + ((v >> 1) << 4) + ((v & 1) << 2) + (hf << 3);
            av[v] = *(const int*)&rows[ra][ka];
            int kbb = kb + ((v >> 1) << 4) + ((v & 1) << 2) + (hf << 3);
            bv[v] = *(const int*)&rows[rb][kbb];
          }
          acc = __builtin_amdgcn_wmma_i32_16x16x64_iu8(true, av, true, bv, acc,
                                                       false, false);
        }
        #pragma unroll
        for (int r = 0; r < 8; ++r) {
          int i = ti * 16 + r + hf * 8;
          int j = tj * 16 + m15;
          int d = norms[i] + norms[j] - 2 * acc[r];
          if (d == 0)
            atomicOr(&eqm[i][j >> 5], 1u << (j & 31));
        }
      }
    }
    __syncthreads();

    for (int i = tid; i < C; i += NTHREADS) {
      int ri = (i < B * Kc) ? (i / Kc) : (B + (i - B * Kc));
      float maxb = MINF, maxn = MINF;
      bool first = true;
      for (int j = 0; j < C; ++j) {
        int rj = (j < B * Kc) ? (j / Kc) : (B + (j - B * Kc));
        if ((eqm[ri][rj >> 5] >> (rj & 31)) & 1u) {
          maxb = fmaxf(maxb, cpb[j]);
          maxn = fmaxf(maxn, cpnb[j]);
          if (j < i) first = false;
        }
      }
      float sb = 0.0f, sn = 0.0f;
      for (int j = 0; j < C; ++j) {
        int rj = (j < B * Kc) ? (j / Kc) : (B + (j - B * Kc));
        if ((eqm[ri][rj >> 5] >> (rj & 31)) & 1u) {
          sb += __expf(cpb[j] - maxb);
          sn += __expf(cpnb[j] - maxn);
        }
      }
      float mb = (maxb < -1e29f) ? NEGF : maxb + __logf(sb);
      float mn = (maxn < -1e29f) ? NEGF : maxn + __logf(sn);
      mpb[i] = mb; mpnb[i] = mn;
      keyv[i] = first ? laddexp(mb, mn) : DUPF;
    }
    __syncthreads();

    if (tid == 0) {
      for (int bb = 0; bb < B; ++bb) {
        float best = MINF; int bi = 0;
        for (int c = 0; c < C; ++c)
          if (keyv[c] > best) { best = keyv[c]; bi = c; }
        sel[bb] = bi;
        keyv[bi] = MINF;
      }
    }
    __syncthreads();

    for (int w = tid; w < B * TW; w += NTHREADS) {
      int b = w / TW, wd = w % TW;
      int c = sel[b];
      int rc = (c < B * Kc) ? (c / Kc) : (B + (c - B * Kc));
      ((int*)cur[b])[wd] = ((const int*)rows[rc])[wd];
    }
    if (tid < B) {
      int c = sel[tid];
      beam_pb[tid]  = mpb[c];
      beam_pnb[tid] = mpnb[c];
      beam_len[tid] = clen[c];
    }
    __syncthreads();
  }

  if (tid < B) out[tid] = laddexp(beam_pb[tid], beam_pnb[tid]);
  int* iout = (int*)out;
  for (int i = tid; i < B * T_LEN; i += NTHREADS) {
    int b = i / T_LEN, tt = i % T_LEN;
    iout[B + i] = (int)cur[b][tt];
  }
  if (tid < B) iout[B + B * T_LEN + tid] = beam_len[tid];
}

extern "C" void kernel_launch(void* const* d_in, const int* in_sizes, int n_in,
                              void* d_out, int out_size, void* d_ws, size_t ws_size,
                              hipStream_t stream) {
  (void)in_sizes; (void)n_in; (void)out_size; (void)d_ws; (void)ws_size;
  const float* logits = (const float*)d_in[0];
  const int*   beam   = (const int*)d_in[1];
  const int*   blank  = (const int*)d_in[2];
  float*       out    = (float*)d_out;
  ctc_prefix_beam_kernel<<<dim3(1), dim3(NTHREADS), 0, stream>>>(logits, beam, blank, out);
}
